// Attention_75788992906123
// MI455X (gfx1250) — matrix operand order repaired and checked by reading, not compiled, not run on hardware
//
#include <hip/hip_runtime.h>

// ---------------------------------------------------------------------------
// CDNA5 (gfx1250) channel-attention pipeline.
// - All GEMM stages: v_wmma_f32_16x16x32_bf16 (ISA lane layouts, CDNA5 ISA)
// - q*q^T reduction: TDM tensor_load_to_lds double-buffered into LDS,
//   s_wait_tensorcnt + workgroup barriers, ds_load_b128 operand fetch.
// ---------------------------------------------------------------------------

typedef __attribute__((ext_vector_type(16))) __bf16 v16bf;
typedef __attribute__((ext_vector_type(8)))  __bf16 v8bf;
typedef __attribute__((ext_vector_type(8)))  float  v8f;
typedef __attribute__((ext_vector_type(4)))  unsigned int u32x4;
typedef __attribute__((ext_vector_type(4)))  int    i32x4;
typedef __attribute__((ext_vector_type(8)))  int    i32x8;

#define NPIX   16384     // H*W = 128*128
#define HPIX   128
#define WPIX   128
#define DIMC   192
#define QKVC   384
#define NHEAD  4
#define HEADC  48
#define NBATCH 8
#define KCHUNK 256       // TDM tile K length (48x256 bf16 = 24 KB per buffer)

#if defined(__AMDGCN__) && __has_builtin(__builtin_amdgcn_tensor_load_to_lds)
#define HAVE_TDM 1
#else
#define HAVE_TDM 0
#endif

static __device__ __forceinline__ v8f wmma_bf16(v16bf a, v16bf b, v8f c) {
    // D = A(16x32 bf16) * B(32x16 bf16) + C(16x16 f32)
    return __builtin_amdgcn_wmma_f32_16x16x32_bf16(false, a, false, b,
                                                   (short)0, c, false, false);
}

// A-matrix element e : K = 16*half + e (contiguous).
static __device__ __forceinline__ v16bf load_a_bf(const __bf16* p) {
    v8bf lo = *(const v8bf*)(p);
    v8bf hi = *(const v8bf*)(p + 8);
    v16bf a;
    #pragma unroll
    for (int e = 0; e < 8; ++e) { a[e] = lo[e]; a[8 + e] = hi[e]; }
    return a;
}

// Same A layout converting from f32 (weights): four aligned float4 loads.
static __device__ __forceinline__ v16bf cvt_a_f32(const float* p) {
    float4 f0 = *(const float4*)(p);
    float4 f1 = *(const float4*)(p + 4);
    float4 f2 = *(const float4*)(p + 8);
    float4 f3 = *(const float4*)(p + 12);
    v16bf a;
    a[0]  = (__bf16)f0.x; a[1]  = (__bf16)f0.y; a[2]  = (__bf16)f0.z; a[3]  = (__bf16)f0.w;
    a[4]  = (__bf16)f1.x; a[5]  = (__bf16)f1.y; a[6]  = (__bf16)f1.z; a[7]  = (__bf16)f1.w;
    a[8]  = (__bf16)f2.x; a[9]  = (__bf16)f2.y; a[10] = (__bf16)f2.z; a[11] = (__bf16)f2.w;
    a[12] = (__bf16)f3.x; a[13] = (__bf16)f3.y; a[14] = (__bf16)f3.z; a[15] = (__bf16)f3.w;
    return a;
}

// TDM: DMA a 48 x tile_k bf16 tile (row stride NPIX elements) into LDS.
// D# packing per CDNA5 ISA (groups 0 and 1).
static __device__ __forceinline__ void tdm_load_tile(const void* gsrc, void* lds_dst,
                                                     int tile_k)
{
#if HAVE_TDM
    unsigned long long ga = (unsigned long long)(uintptr_t)gsrc;
    unsigned int lds_off  = (unsigned int)(uintptr_t)lds_dst; // flat LDS addr[31:0]
    u32x4 g0;
    g0.x = 1u;                                                 // count=1, user mode
    g0.y = lds_off;                                            // lds_addr
    g0.z = (unsigned int)ga;                                   // global_addr[31:0]
    g0.w = (unsigned int)((ga >> 32) & 0x01FFFFFFu) | (2u << 30); // addr hi | type=2
    i32x8 g1;
    g1[0] = (int)(1u << 16);                        // wg_mask=0, data_size=1 (2B)
    g1[1] = (int)((NPIX & 0xFFFF) << 16);           // tensor_dim0[15:0]
    g1[2] = (int)((NPIX >> 16) | (HEADC << 16));    // dim0 hi | tensor_dim1 lo
    g1[3] = (int)((tile_k & 0xFFFF) << 16);         // dim1 hi=0 | tile_dim0
    g1[4] = HEADC;                                  // tile_dim1=48, tile_dim2=0
    g1[5] = NPIX;                                   // tensor_dim0_stride[31:0]
    g1[6] = 0;                                      // stride hi | dim1_stride lo
    g1[7] = 0;
    i32x4 z4 = {0, 0, 0, 0};
#if __clang_major__ >= 23
    i32x8 z8 = {};
    __builtin_amdgcn_tensor_load_to_lds(g0, g1, z4, z4, z8, 0);
#else
    __builtin_amdgcn_tensor_load_to_lds(g0, g1, z4, z4, 0);
#endif
#else
    (void)gsrc; (void)lds_dst; (void)tile_k;
#endif
}

static __device__ __forceinline__ void wait_tensor0()
{
#if defined(__AMDGCN__) && __has_builtin(__builtin_amdgcn_s_wait_tensorcnt)
    __builtin_amdgcn_s_wait_tensorcnt(0);
#endif
}

// ---------------------------------------------------------------------------
// 0) Transpose-convert: x [b][192][16384] f32 -> xT [b][16384][192] bf16.
// ---------------------------------------------------------------------------
__global__ __launch_bounds__(256)
void k_xpose(const float* __restrict__ in, __bf16* __restrict__ out)
{
    __shared__ float tile[32][33];
    const int b  = blockIdx.z;
    const int c0 = blockIdx.y * 32;
    const int n0 = blockIdx.x * 32;
    const float* ib = in  + (size_t)b * DIMC * NPIX;
    __bf16*      ob = out + (size_t)b * NPIX * DIMC;
    const int tx = threadIdx.x, ty = threadIdx.y;   // (32, 8)
    #pragma unroll
    for (int i = 0; i < 4; ++i)
        tile[ty + 8 * i][tx] = ib[(size_t)(c0 + ty + 8 * i) * NPIX + n0 + tx];
    __syncthreads();
    #pragma unroll
    for (int i = 0; i < 4; ++i)
        ob[(size_t)(n0 + ty + 8 * i) * DIMC + c0 + tx] = (__bf16)tile[tx][ty + 8 * i];
}

// ---------------------------------------------------------------------------
// 1) qkv = w_qkv[384x192] * x[b][192x16384]. Wave: 64M x 32N, 8 WMMA/K-step.
// ---------------------------------------------------------------------------
__global__ __launch_bounds__(32)
void k_qkv_gemm(const __bf16* __restrict__ xT, const float* __restrict__ wqkv,
                float* __restrict__ qkv)
{
    const int lane = threadIdx.x, half = lane >> 4, l16 = lane & 15;
    const int n0 = blockIdx.x * 32;
    const int o0 = blockIdx.y * 64;
    const int b  = blockIdx.z;
    const __bf16* xb = xT + (size_t)b * NPIX * DIMC;

    v8f acc[4][2] = {};
    for (int k0 = 0; k0 < DIMC; k0 += 32) {
        v16bf a[4];
        #pragma unroll
        for (int mt = 0; mt < 4; ++mt)
            a[mt] = cvt_a_f32(wqkv + (o0 + mt * 16 + l16) * DIMC + k0 + 16 * half);
        v16bf bt[2];
        #pragma unroll
        for (int nt = 0; nt < 2; ++nt)
            bt[nt] = *(const v16bf*)(xb + (size_t)(n0 + nt * 16 + l16) * DIMC
                                        + k0 + 16 * half);
        #pragma unroll
        for (int mt = 0; mt < 4; ++mt)
            #pragma unroll
            for (int nt = 0; nt < 2; ++nt)
                acc[mt][nt] = wmma_bf16(a[mt], bt[nt], acc[mt][nt]);
    }
    float* yb = qkv + (size_t)b * QKVC * NPIX;
    #pragma unroll
    for (int mt = 0; mt < 4; ++mt)
        #pragma unroll
        for (int nt = 0; nt < 2; ++nt)
            #pragma unroll
            for (int r = 0; r < 8; ++r)
                yb[(size_t)(o0 + mt * 16 + r + 8 * half) * NPIX + n0 + nt * 16 + l16] =
                    acc[mt][nt][r];
}

// ---------------------------------------------------------------------------
// 2) 3x3 depthwise conv (SAME, zero pad). q-half -> f32, v-half -> bf16.
// ---------------------------------------------------------------------------
__global__ __launch_bounds__(256)
void k_dwconv(const float* __restrict__ qkv, const float* __restrict__ wdw,
              float* __restrict__ q_dw, __bf16* __restrict__ v_bf)
{
    size_t idx = (size_t)blockIdx.x * blockDim.x + threadIdx.x;
    int n  = (int)(idx & (NPIX - 1));
    int bc = (int)(idx >> 14);
    int ch = bc % QKVC;
    int b  = bc / QKVC;
    int y = n >> 7, xw = n & (WPIX - 1);

    const float* src = qkv + ((size_t)b * QKVC + ch) * NPIX;
    const float* wf  = wdw + ch * 9;
    float s = 0.f;
    #pragma unroll
    for (int dy = -1; dy <= 1; ++dy) {
        int yy = y + dy;
        if (yy < 0 || yy >= HPIX) continue;
        #pragma unroll
        for (int dx = -1; dx <= 1; ++dx) {
            int xx = xw + dx;
            if (xx < 0 || xx >= WPIX) continue;
            s += wf[(dy + 1) * 3 + (dx + 1)] * src[yy * WPIX + xx];
        }
    }
    if (ch < DIMC) q_dw[((size_t)b * DIMC + ch) * NPIX + n] = s;
    else           v_bf[((size_t)b * DIMC + (ch - DIMC)) * NPIX + n] = (__bf16)s;
}

// ---------------------------------------------------------------------------
// 3) L2-normalize each (b,c) row of q over the 16384 spatial dim -> bf16.
// ---------------------------------------------------------------------------
__global__ __launch_bounds__(256)
void k_norm(const float* __restrict__ q_dw, __bf16* __restrict__ q_bf)
{
    const int row = blockIdx.x;                 // b*192 + c
    const float* p = q_dw + (size_t)row * NPIX;
    __shared__ float red[256];
    float s = 0.f;
    for (int i = threadIdx.x; i < NPIX; i += 256) { float v = p[i]; s += v * v; }
    red[threadIdx.x] = s;
    __syncthreads();
    for (int off = 128; off > 0; off >>= 1) {
        if ((int)threadIdx.x < off) red[threadIdx.x] += red[threadIdx.x + off];
        __syncthreads();
    }
    float inv = rsqrtf(red[0]);
    __bf16* q = q_bf + (size_t)row * NPIX;
    for (int i = threadIdx.x; i < NPIX; i += 256) q[i] = (__bf16)(p[i] * inv);
}

// ---------------------------------------------------------------------------
// 4a) Partial q*q^T via TDM double-buffered LDS pipeline.
//     Block = 3 waves (one per 16-row M-tile) sharing one 48 x 256 LDS tile.
//     Each block covers a K-range of 1024 (4 chunks); grid (32 bh, 16 kr).
//     f32 accumulation via global_atomic_add_f32.
// ---------------------------------------------------------------------------
__global__ __launch_bounds__(96)
void k_attn_tdm(const __bf16* __restrict__ q_bf, float* __restrict__ attn_f32)
{
    __shared__ __attribute__((aligned(64))) __bf16 smem[2][HEADC * KCHUNK];
    const int tid  = threadIdx.x;
    const int lane = tid & 31, wv = tid >> 5;       // wv = M-tile (0..2)
    const int half = (lane >> 4) & 1, l16 = lane & 15;
    const int bh = blockIdx.x;                      // 0..31
    const int kr = blockIdx.y;                      // 0..15
    const int b = bh >> 2, h = bh & 3;
    const __bf16* qh = q_bf + ((size_t)b * DIMC + h * HEADC) * NPIX;
    const int krange = kr * (4 * KCHUNK);

    // Fill / prefetch one chunk into an LDS buffer.
    auto issue = [&](int chunk, int buf) {
#if HAVE_TDM
        if (wv == 0)
            tdm_load_tile(qh + krange + chunk * KCHUNK, &smem[buf][0], KCHUNK);
#else
        for (int i = tid; i < HEADC * (KCHUNK / 8); i += 96) {
            int rr = i / (KCHUNK / 8), cc = (i % (KCHUNK / 8)) * 8;
            *(v8bf*)&smem[buf][rr * KCHUNK + cc] =
                *(const v8bf*)(qh + (size_t)rr * NPIX + krange + chunk * KCHUNK + cc);
        }
#endif
    };
    auto wait_fill = [&]() {
#if HAVE_TDM
        if (wv == 0) wait_tensor0();                // TDM done writing LDS
#endif
        __syncthreads();                            // publish to all 3 waves
    };

    issue(0, 0);
    wait_fill();

    v8f acc[3] = {};
    for (int c = 0; c < 4; ++c) {
        const int buf = c & 1;
        if (c + 1 < 4) issue(c + 1, buf ^ 1);       // overlap DMA with compute
        const __bf16* sm = &smem[buf][0];
        #pragma unroll
        for (int k0 = 0; k0 < KCHUNK; k0 += 32) {
            v16bf a = load_a_bf(sm + (wv * 16 + l16) * KCHUNK + k0 + 16 * half);
            #pragma unroll
            for (int t = 0; t < 3; ++t) {
                v16bf bt = *(const v16bf*)(sm + (t * 16 + l16) * KCHUNK
                                              + k0 + 16 * half);
                acc[t] = wmma_bf16(a, bt, acc[t]);
            }
        }
        wait_fill();
    }

    float* ab = attn_f32 + (size_t)bh * HEADC * HEADC;
    #pragma unroll
    for (int t = 0; t < 3; ++t)
        #pragma unroll
        for (int r = 0; r < 8; ++r)
            atomicAdd(&ab[(wv * 16 + r + 8 * half) * HEADC + t * 16 + l16],
                      acc[t][r]);
}

// ---------------------------------------------------------------------------
// 4b) softmax(attn * temp[h]) rows; emit zero-padded 64x64 bf16 tile.
// ---------------------------------------------------------------------------
__global__ __launch_bounds__(64)
void k_softmax(const float* __restrict__ attn_f32, const float* __restrict__ temp,
               __bf16* __restrict__ attn_bf)
{
    const int bh = blockIdx.x, h = bh & 3;
    const int row = threadIdx.x;
    if (row >= HEADC) return;
    const float ts = temp[h];
    const float* pr = attn_f32 + ((size_t)bh * HEADC + row) * HEADC;
    float v[HEADC];
    float m = -1e30f;
    #pragma unroll
    for (int j = 0; j < HEADC; ++j) { v[j] = pr[j] * ts; m = fmaxf(m, v[j]); }
    float s = 0.f;
    #pragma unroll
    for (int j = 0; j < HEADC; ++j) { v[j] = __expf(v[j] - m); s += v[j]; }
    const float rs = 1.f / s;
    __bf16* po = attn_bf + ((size_t)bh * 64 + row) * 64;
    #pragma unroll
    for (int j = 0; j < HEADC; ++j) po[j] = (__bf16)(v[j] * rs);
    #pragma unroll
    for (int j = HEADC; j < 64; ++j) po[j] = (__bf16)0.f;   // pad cols 48..63
}

// ---------------------------------------------------------------------------
// 5) out = attn * v per (b,h); store TRANSPOSED outT[b][n][c] bf16
//    (per-lane packed 16B v8bf stores; feeds k_proj with contiguous B).
// ---------------------------------------------------------------------------
__global__ __launch_bounds__(32)
void k_av(const __bf16* __restrict__ attn, const __bf16* __restrict__ v_bf,
          __bf16* __restrict__ outT)
{
    const int lane = threadIdx.x, half = lane >> 4, l16 = lane & 15;
    const int n0 = blockIdx.x * 64;
    const int mt = blockIdx.y;                  // 0..2
    const int bh = blockIdx.z;                  // 0..31
    const int b = bh >> 2, h = bh & 3;
    const __bf16* A  = attn + (size_t)bh * 64 * 64;
    const __bf16* vb = v_bf + (size_t)b * DIMC * NPIX;

    v8f acc[4] = {};
    #pragma unroll
    for (int k0 = 0; k0 < 64; k0 += 32) {
        v16bf a = load_a_bf(A + (mt * 16 + l16) * 64 + k0 + 16 * half);
        #pragma unroll
        for (int t = 0; t < 4; ++t) {
            v16bf bt;
            #pragma unroll
            for (int e = 0; e < 16; ++e) {
                int kk = k0 + e + 16 * half;                        // 0..63
                int ch = h * HEADC + (kk < HEADC ? kk : HEADC - 1); // clamp; A==0 there
                bt[e] = vb[(size_t)ch * NPIX + n0 + t * 16 + l16];
            }
            acc[t] = wmma_bf16(a, bt, acc[t]);
        }
    }
    __bf16* ob = outT + (size_t)b * NPIX * DIMC;
    #pragma unroll
    for (int t = 0; t < 4; ++t) {
        v8bf pk;
        #pragma unroll
        for (int r = 0; r < 8; ++r) pk[r] = (__bf16)acc[t][r];
        *(v8bf*)(ob + (size_t)(n0 + t * 16 + l16) * DIMC
                    + h * HEADC + mt * 16 + 8 * half) = pk;
    }
}

// ---------------------------------------------------------------------------
// 6) y = w_proj[192x192] * out[b][192x16384]; B from outT (contiguous loads).
// ---------------------------------------------------------------------------
__global__ __launch_bounds__(32)
void k_proj(const __bf16* __restrict__ outT, const float* __restrict__ wp,
            float* __restrict__ y)
{
    const int lane = threadIdx.x, half = lane >> 4, l16 = lane & 15;
    const int n0 = blockIdx.x * 32;
    const int o0 = blockIdx.y * 64;
    const int b  = blockIdx.z;
    const __bf16* ob = outT + (size_t)b * NPIX * DIMC;

    v8f acc[4][2] = {};
    for (int k0 = 0; k0 < DIMC; k0 += 32) {
        v16bf a[4];
        #pragma unroll
        for (int mt = 0; mt < 4; ++mt)
            a[mt] = cvt_a_f32(wp + (o0 + mt * 16 + l16) * DIMC + k0 + 16 * half);
        v16bf bt[2];
        #pragma unroll
        for (int nt = 0; nt < 2; ++nt)
            bt[nt] = *(const v16bf*)(ob + (size_t)(n0 + nt * 16 + l16) * DIMC
                                        + k0 + 16 * half);
        #pragma unroll
        for (int mt = 0; mt < 4; ++mt)
            #pragma unroll
            for (int nt = 0; nt < 2; ++nt)
                acc[mt][nt] = wmma_bf16(a[mt], bt[nt], acc[mt][nt]);
    }
    float* yb = y + (size_t)b * DIMC * NPIX;
    #pragma unroll
    for (int mt = 0; mt < 4; ++mt)
        #pragma unroll
        for (int nt = 0; nt < 2; ++nt)
            #pragma unroll
            for (int r = 0; r < 8; ++r)
                yb[(size_t)(o0 + mt * 16 + r + 8 * half) * NPIX + n0 + nt * 16 + l16] =
                    acc[mt][nt][r];
}

// ---------------------------------------------------------------------------
extern "C" void kernel_launch(void* const* d_in, const int* in_sizes, int n_in,
                              void* d_out, int out_size, void* d_ws, size_t ws_size,
                              hipStream_t stream)
{
    (void)in_sizes; (void)n_in; (void)out_size; (void)ws_size;
    const float* x      = (const float*)d_in[0];
    const float* w_qkv  = (const float*)d_in[1];
    const float* w_dw   = (const float*)d_in[2];
    const float* w_proj = (const float*)d_in[3];
    const float* temp   = (const float*)d_in[4];
    float* y = (float*)d_out;

    char* ws = (char*)d_ws;
    const size_t SZ_QKV  = (size_t)NBATCH * QKVC * NPIX * sizeof(float);   // 201.3 MB
    const size_t SZ_QDW  = (size_t)NBATCH * DIMC * NPIX * sizeof(float);   // 100.7 MB
    const size_t SZ_VBF  = (size_t)NBATCH * DIMC * NPIX * sizeof(__bf16);  // 50.3 MB
    const size_t SZ_XT   = SZ_VBF;                                         // 50.3 MB
    const size_t SZ_AF32 = (size_t)NBATCH * NHEAD * HEADC * HEADC * sizeof(float);

    float*  qkv      = (float*)ws;
    float*  q_dw     = (float*)(ws + SZ_QKV);
    __bf16* v_bf     = (__bf16*)(ws + SZ_QKV + SZ_QDW);
    __bf16* xT       = (__bf16*)(ws + SZ_QKV + SZ_QDW + SZ_VBF);
    float*  attn_f32 = (float*)(ws + SZ_QKV + SZ_QDW + SZ_VBF + SZ_XT);
    __bf16* attn_bf  = (__bf16*)(ws + SZ_QKV + SZ_QDW + SZ_VBF + SZ_XT + SZ_AF32);
    // Aliases into the qkv region — safe: qkv is fully consumed by k_dwconv
    // before either alias is written (stream-ordered):
    __bf16* q_bf = (__bf16*)ws;              // [8][192][16384] bf16
    __bf16* outT = (__bf16*)(ws + SZ_VBF);   // [8][16384][192] bf16

    k_xpose<<<dim3(NPIX / 32, DIMC / 32, NBATCH), dim3(32, 8), 0, stream>>>(x, xT);
    k_qkv_gemm<<<dim3(NPIX / 32, QKVC / 64, NBATCH), 32, 0, stream>>>(xT, w_qkv, qkv);
    k_dwconv<<<(NBATCH * QKVC * NPIX) / 256, 256, 0, stream>>>(qkv, w_dw, q_dw, v_bf);
    k_norm<<<NBATCH * DIMC, 256, 0, stream>>>(q_dw, q_bf);
    hipMemsetAsync(attn_f32, 0, SZ_AF32, stream);
    k_attn_tdm<<<dim3(NBATCH * NHEAD, 16), 96, 0, stream>>>(q_bf, attn_f32);
    k_softmax<<<NBATCH * NHEAD, 64, 0, stream>>>(attn_f32, temp, attn_bf);
    k_av<<<dim3(NPIX / 64, 3, NBATCH * NHEAD), 32, 0, stream>>>(attn_bf, v_bf, outT);
    k_proj<<<dim3(NPIX / 32, DIMC / 64, NBATCH), 32, 0, stream>>>(outT, w_proj, y);
}
